// ContextualBlock_10075993276865
// MI455X (gfx1250) — matrix operand order repaired and checked by reading, not compiled, not run on hardware
//
#include <hip/hip_runtime.h>
#include <math.h>

// ---------------------------------------------------------------------------
// Contextual attention block for MI455X (gfx1250), wave32 + WMMA bf16
// + async global->LDS double buffering (ASYNCcnt path) when available.
// B=4, C=64, H=W=64, K=3, STRIDE=2 -> L=961 (pad 1024), d=C*K*K=576, M=4096.
// ---------------------------------------------------------------------------

typedef __attribute__((ext_vector_type(16))) __bf16 v16bf;
typedef __attribute__((ext_vector_type(8)))  __bf16 v8bf;
typedef __attribute__((ext_vector_type(8)))  float  v8f;
typedef int v4i_vs __attribute__((vector_size(16)));   // matches builtin param

#define BATCH 4
#define CCH   64
#define HH    64
#define WW_   64
#define DD    576      // C*K*K
#define DD2   640      // DD padded to a multiple of 128 (GEMM2 N tile)
#define MM    4096     // H*W queries
#define LREAL 961      // 31*31 keys
#define LPAD  1024
#define LDSP  40       // LDS row pitch in bf16 elems (80B; stride-20 banks)
#define TSZ   (128 * LDSP)   // one 128x32 tile in LDS (elems)

#if defined(__gfx1250__) && __has_builtin(__builtin_amdgcn_global_load_async_to_lds_b128)
#define HAVE_ASYNC_LDS 1
#else
#define HAVE_ASYNC_LDS 0
#endif

__device__ __forceinline__ void async_b128(const __bf16* g, __bf16* l) {
#if HAVE_ASYNC_LDS
    // builtin signature (from hipcc diagnostic): arg0 = AS(1) int4*, arg1 = AS(3) int4*
    __builtin_amdgcn_global_load_async_to_lds_b128(
        (__attribute__((address_space(1))) v4i_vs*)g,
        (__attribute__((address_space(3))) v4i_vs*)l, 0, 0);
#else
    (void)g; (void)l;
#endif
}

__device__ __forceinline__ void wait_async0() {
#if defined(__gfx1250__)
#if __has_builtin(__builtin_amdgcn_s_wait_asynccnt)
    __builtin_amdgcn_s_wait_asynccnt(0);
#else
    asm volatile("s_wait_asynccnt 0x0" ::: "memory");
#endif
#endif
}

// ------------------------- prep: query matrix (bf16) -----------------------
// Q[b][m][d] = fg_in patch (k=3, s=1, pad=1), m = h*64+w, d = c*9+i*3+j
__global__ void k_build_q(const float* __restrict__ fg, __bf16* __restrict__ Q) {
    int idx = blockIdx.x * 256 + threadIdx.x;
    if (idx >= BATCH * MM * DD) return;
    int d = idx % DD;
    int m = (idx / DD) & (MM - 1);
    int b = idx / (DD * MM);
    int c = d / 9, r = d % 9, i = r / 3, j = r % 3;
    int h = m >> 6, w = m & 63;
    int hh = h + i - 1, ww = w + j - 1;
    float v = 0.f;
    if (hh >= 0 && hh < HH && ww >= 0 && ww < WW_)
        v = fg[(((size_t)b * CCH + c) * HH + hh) * WW_ + ww];
    Q[idx] = (__bf16)v;
}

// ------------------ prep: key matrix + transposed copy ---------------------
// UBr[b][n][d<576] (GEMM1 B-operand), UBt[b][d<640][n] (GEMM2 B-op, pad rows 0)
__global__ void k_build_k(const float* __restrict__ bg, const float* __restrict__ mask,
                          __bf16* __restrict__ UBr, __bf16* __restrict__ UBt) {
    int idx = blockIdx.x * 256 + threadIdx.x;
    if (idx >= BATCH * LPAD * DD2) return;
    int d = idx % DD2;
    int n = (idx / DD2) & (LPAD - 1);
    int b = idx / (DD2 * LPAD);
    float v = 0.f;
    if (n < LREAL && d < DD) {
        int ly = n / 31, lx = n % 31;
        int c = d / 9, r = d % 9, i = r / 3, j = r % 3;
        int h = ly * 2 + i, w = lx * 2 + j;
        float mv = mask[(size_t)b * HH * WW_ + h * WW_ + w];
        v = bg[(((size_t)b * CCH + c) * HH + h) * WW_ + w] * mv;
    }
    if (d < DD) UBr[((size_t)b * LPAD + n) * DD + d] = (__bf16)v;
    UBt[((size_t)b * DD2 + d) * LPAD + n] = (__bf16)v;
}

// ----------------------------- row norms -----------------------------------
__global__ void k_qnorm(const __bf16* __restrict__ Q, float* __restrict__ wwd) {
    int idx = blockIdx.x * 256 + threadIdx.x;
    if (idx >= BATCH * MM) return;
    const __bf16* p = Q + (size_t)idx * DD;
    float s = 0.f;
    for (int d = 0; d < DD; ++d) { float v = (float)p[d]; s += v * v; }
    wwd[idx] = s;
}

__global__ void k_knorm(const __bf16* __restrict__ UBr, float* __restrict__ k1d) {
    int idx = blockIdx.x * 256 + threadIdx.x;
    if (idx >= BATCH * LPAD) return;
    const __bf16* p = UBr + (size_t)idx * DD;
    float s = 0.f;
    for (int d = 0; d < DD; ++d) { float v = (float)p[d]; s += v * v; }
    k1d[idx] = s;
}

// ----------------------- WMMA bf16 TN GEMM ---------------------------------
// C[m][n] = sum_k A[m][k] * B[n][k]; A: MxK row-major, B: NxK row-major.
// Block: 256 thr = 8 waves (4M x 2N); tile 128(M) x 128(N); wave = 32x64 =
// 2x4 WMMA 16x16x32 bf16 tiles. Double-buffered LDS fed by async-to-LDS.
// mode 1: C = k1d[n] + wwd[m] - 2*acc   (squared L2 distance epilogue)
// mode 0: C = acc
__global__ void __launch_bounds__(256)
gemm_tn_wmma(const __bf16* __restrict__ A, const __bf16* __restrict__ Bm,
             float* __restrict__ C,
             const float* __restrict__ k1d, const float* __restrict__ wwd,
             int K, int ldA, int ldB, int ldC,
             size_t sA, size_t sB, size_t sC, int mode) {
    const int b    = blockIdx.z;
    const int tid  = threadIdx.x;
    const int lane = tid & 31;
    const int wid  = tid >> 5;
    const int wm   = wid & 3;   // 4 waves along M (32 rows each)
    const int wn   = wid >> 2;  // 2 waves along N (64 cols each)

    const __bf16* Ab = A  + (size_t)b * sA + (size_t)blockIdx.x * 128 * ldA;
    const __bf16* Bb = Bm + (size_t)b * sB + (size_t)blockIdx.y * 128 * ldB;
    float*        Cb = C  + (size_t)b * sC;

    __shared__ __align__(16) __bf16 As[2 * TSZ];  // 2 x 128x32 (pitch 40)
    __shared__ __align__(16) __bf16 Bs[2 * TSZ];

    const v8f zero = {0.f, 0.f, 0.f, 0.f, 0.f, 0.f, 0.f, 0.f};
    v8f acc[2][4] = {{zero, zero, zero, zero}, {zero, zero, zero, zero}};

    const int kgA = (lane >> 4) << 3;   // A frag K half offset: 0 / 8
    const int kbB = (lane >> 4) << 3;   // B frag K half offset: 0 / 8
    const int nk  = K >> 5;

    // chunk mapping: 512 16B-chunks per 128x32 tile -> 2 per thread
    const int r0 = tid >> 2,          c0 = tid & 3;          // chunk tid
    const int r1 = (tid + 256) >> 2,  c1 = tid & 3;          // chunk tid+256

#if HAVE_ASYNC_LDS
    // ---- issue tile 0 into buffer 0 ----
    {
        async_b128(Ab + (size_t)r0 * ldA + (c0 << 3), &As[r0 * LDSP + (c0 << 3)]);
        async_b128(Ab + (size_t)r1 * ldA + (c1 << 3), &As[r1 * LDSP + (c1 << 3)]);
        async_b128(Bb + (size_t)r0 * ldB + (c0 << 3), &Bs[r0 * LDSP + (c0 << 3)]);
        async_b128(Bb + (size_t)r1 * ldB + (c1 << 3), &Bs[r1 * LDSP + (c1 << 3)]);
    }
#endif

    for (int kk = 0; kk < nk; ++kk) {
        const int cur  = kk & 1;
        const int k0   = kk << 5;
#if HAVE_ASYNC_LDS
        wait_async0();          // my async writes of tile kk have landed in LDS
        __syncthreads();        // all waves' writes visible; prev compute done
        if (kk + 1 < nk) {      // prefetch tile kk+1 into the other buffer
            const int kn = k0 + 32, o = (cur ^ 1) * TSZ;
            async_b128(Ab + (size_t)r0 * ldA + kn + (c0 << 3), &As[o + r0 * LDSP + (c0 << 3)]);
            async_b128(Ab + (size_t)r1 * ldA + kn + (c1 << 3), &As[o + r1 * LDSP + (c1 << 3)]);
            async_b128(Bb + (size_t)r0 * ldB + kn + (c0 << 3), &Bs[o + r0 * LDSP + (c0 << 3)]);
            async_b128(Bb + (size_t)r1 * ldB + kn + (c1 << 3), &Bs[o + r1 * LDSP + (c1 << 3)]);
        }
        const __bf16* Asb = &As[cur * TSZ];
        const __bf16* Bsb = &Bs[cur * TSZ];
#else
        // fallback: sync staging through VGPRs, single buffer
        v8bf ra0 = *reinterpret_cast<const v8bf*>(Ab + (size_t)r0 * ldA + k0 + (c0 << 3));
        v8bf ra1 = *reinterpret_cast<const v8bf*>(Ab + (size_t)r1 * ldA + k0 + (c1 << 3));
        v8bf rb0 = *reinterpret_cast<const v8bf*>(Bb + (size_t)r0 * ldB + k0 + (c0 << 3));
        v8bf rb1 = *reinterpret_cast<const v8bf*>(Bb + (size_t)r1 * ldB + k0 + (c1 << 3));
        if (kk + 1 < nk) {
            __builtin_prefetch(Ab + (size_t)r0 * ldA + k0 + 32, 0, 1);
            __builtin_prefetch(Bb + (size_t)r0 * ldB + k0 + 32, 0, 1);
        }
        __syncthreads();
        *reinterpret_cast<v8bf*>(&As[r0 * LDSP + (c0 << 3)]) = ra0;
        *reinterpret_cast<v8bf*>(&As[r1 * LDSP + (c1 << 3)]) = ra1;
        *reinterpret_cast<v8bf*>(&Bs[r0 * LDSP + (c0 << 3)]) = rb0;
        *reinterpret_cast<v8bf*>(&Bs[r1 * LDSP + (c1 << 3)]) = rb1;
        __syncthreads();
        const __bf16* Asb = As;
        const __bf16* Bsb = Bs;
#endif
        // ---- fragments from LDS per ISA lane layouts ----
        v16bf afr[2], bfr[4];
#pragma unroll
        for (int mt = 0; mt < 2; ++mt) {
            int row = (wm << 5) + (mt << 4) + (lane & 15);
            v8bf lo = *reinterpret_cast<const v8bf*>(&Asb[row * LDSP + kgA]);
            v8bf hi = *reinterpret_cast<const v8bf*>(&Asb[row * LDSP + 16 + kgA]);
            afr[mt] = __builtin_shufflevector(lo, hi, 0,1,2,3,4,5,6,7,8,9,10,11,12,13,14,15);
        }
#pragma unroll
        for (int nt = 0; nt < 4; ++nt) {
            int row = (wn << 6) + (nt << 4) + (lane & 15);
            v8bf lo = *reinterpret_cast<const v8bf*>(&Bsb[row * LDSP + kbB]);
            v8bf hi = *reinterpret_cast<const v8bf*>(&Bsb[row * LDSP + 16 + kbB]);
            bfr[nt] = __builtin_shufflevector(lo, hi, 0,1,2,3,4,5,6,7,8,9,10,11,12,13,14,15);
        }
#pragma unroll
        for (int mt = 0; mt < 2; ++mt)
#pragma unroll
            for (int nt = 0; nt < 4; ++nt)
                acc[mt][nt] = __builtin_amdgcn_wmma_f32_16x16x32_bf16(
                    false, afr[mt], false, bfr[nt], (short)0, acc[mt][nt], false, false);
#if !HAVE_ASYNC_LDS
        __syncthreads();
#endif
    }

    // ---- epilogue: f32 C/D layout -> n = lane&15, m = r + 8*(lane>>4) ----
    const int mBase = blockIdx.x * 128 + (wm << 5);
    const int nBase = blockIdx.y * 128 + (wn << 6);
    const float* k1b = k1d ? (k1d + (size_t)b * LPAD) : nullptr;
    const float* wwb = wwd ? (wwd + (size_t)b * MM)   : nullptr;
#pragma unroll
    for (int mt = 0; mt < 2; ++mt)
#pragma unroll
        for (int nt = 0; nt < 4; ++nt)
#pragma unroll
            for (int r = 0; r < 8; ++r) {
                int m = mBase + (mt << 4) + r + ((lane >> 4) << 3);
                int n = nBase + (nt << 4) + (lane & 15);
                float v = acc[mt][nt][r];
                if (mode) v = k1b[n] + wwb[m] - 2.f * v;
                Cb[(size_t)m * ldC + n] = v;
            }
}

// ---------------- fused mean/std -> -tanh -> softmax row -------------------
__device__ __forceinline__ float blk_reduce(float v, float* red) {
    int tid = threadIdx.x;
    red[tid] = v; __syncthreads();
    for (int s = 128; s > 0; s >>= 1) {
        if (tid < s) red[tid] += red[tid + s];
        __syncthreads();
    }
    float r = red[0]; __syncthreads();
    return r;
}

__global__ void __launch_bounds__(256)
k_softmax(const float* __restrict__ DS1, __bf16* __restrict__ CAb) {
    __shared__ float red[256];
    const int row = blockIdx.x;                         // b*4096 + m
    const float* p = DS1 + (size_t)row * LPAD;
    __bf16*      q = CAb + (size_t)row * LPAD;
    const int tid = threadIdx.x;

    float v[4]; float s = 0.f, s2 = 0.f;
#pragma unroll
    for (int si = 0; si < 4; ++si) {
        int idx = tid + (si << 8);
        float x = (idx < LREAL) ? p[idx] : 0.f;
        v[si] = x;
        if (idx < LREAL) { s += x; s2 += x * x; }
    }
    s  = blk_reduce(s,  red);
    s2 = blk_reduce(s2, red);
    const float mean = s / (float)LREAL;
    const float var  = s2 / (float)LREAL - mean * mean;
    const float inv  = 1.f / (sqrtf(fmaxf(var, 0.f)) + 1e-6f);

    float e[4]; float es = 0.f;
#pragma unroll
    for (int si = 0; si < 4; ++si) {
        int idx = tid + (si << 8);
        if (idx < LREAL) {
            float z = -10.f * tanhf((v[si] - mean) * inv);   // LAMDA * DS2
            e[si] = expf(z);                                 // z in [-10,10]: safe
            es += e[si];
        } else e[si] = 0.f;
    }
    es = blk_reduce(es, red);
    const float rs = 1.f / es;
#pragma unroll
    for (int si = 0; si < 4; ++si) {
        int idx = tid + (si << 8);
        q[idx] = (idx < LREAL) ? (__bf16)(e[si] * rs) : (__bf16)0.f;
    }
}

// ------ fused deconv-gather + blend + concat 1x1 conv + ELU ---------------
// ACL[c,y,x] = (1/9) sum_{i,j} OUT2[(y+1-i)*64+(x+1-j)][c*9+i*3+j] (valid)
// out[co]    = elu( sum_c fw[co][c]*bg_in + fw[co][64+c]*ACLblend + fb[co] )
__global__ void __launch_bounds__(256)
k_final(const float* __restrict__ bg_in, const float* __restrict__ mask,
        const float* __restrict__ OUT2,  const float* __restrict__ fw,
        const float* __restrict__ fb,    float* __restrict__ out) {
    __shared__ float s_acl[CCH * WW_];
    __shared__ float s_bg [CCH * WW_];
    const int blk = blockIdx.x;
    const int b = blk >> 6, y = blk & 63;
    const int tid = threadIdx.x;
    const float* mrow = mask + ((size_t)b * HH + y) * WW_;

    for (int t = tid; t < CCH * WW_; t += 256) {
        int c = t >> 6, x = t & 63;
        float mv   = mrow[x];
        float bgin = bg_in[(((size_t)b * CCH + c) * HH + y) * WW_ + x];
        float sacc = 0.f;
#pragma unroll
        for (int i = 0; i < 3; ++i)
#pragma unroll
            for (int j = 0; j < 3; ++j) {
                int h = y + 1 - i, w = x + 1 - j;
                if (h >= 0 && h < HH && w >= 0 && w < WW_)
                    sacc += OUT2[((size_t)b * MM + h * WW_ + w) * DD2 + c * 9 + i * 3 + j];
            }
        s_bg[t]  = bgin;
        s_acl[t] = bgin * mv + (1.f - mv) * (sacc * (1.f / 9.f));
    }
    __syncthreads();
    for (int t = tid; t < CCH * WW_; t += 256) {
        int co = t >> 6, x = t & 63;
        float acc = fb[co];
        const float* w0 = fw + (size_t)co * (2 * CCH);
        for (int c = 0; c < CCH; ++c)
            acc += w0[c] * s_bg[(c << 6) + x] + w0[CCH + c] * s_acl[(c << 6) + x];
        out[(((size_t)b * CCH + co) * HH + y) * WW_ + x] = acc > 0.f ? acc : expm1f(acc);
    }
}

// ---------------------------------------------------------------------------
extern "C" void kernel_launch(void* const* d_in, const int* in_sizes, int n_in,
                              void* d_out, int out_size, void* d_ws, size_t ws_size,
                              hipStream_t stream) {
    const float* bg_in  = (const float*)d_in[0];
    const float* fg_in  = (const float*)d_in[1];
    const float* mask   = (const float*)d_in[2];
    const float* fuse_w = (const float*)d_in[3];
    const float* fuse_b = (const float*)d_in[4];
    float*       out    = (float*)d_out;

    char* ws = (char*)d_ws;
    size_t off = 0;
    auto carve = [&](size_t bytes) -> void* {
        void* p = ws + off;
        off += (bytes + 255) & ~(size_t)255;
        return p;
    };
    __bf16* Q    = (__bf16*)carve((size_t)BATCH * MM   * DD   * 2); // 18.9 MB
    __bf16* UBr  = (__bf16*)carve((size_t)BATCH * LPAD * DD   * 2); //  4.7 MB
    __bf16* UBt  = (__bf16*)carve((size_t)BATCH * DD2  * LPAD * 2); //  5.2 MB
    float*  k1d  = (float*) carve((size_t)BATCH * LPAD * 4);
    float*  wwd  = (float*) carve((size_t)BATCH * MM   * 4);
    float*  DS1  = (float*) carve((size_t)BATCH * MM   * LPAD * 4); // 67.1 MB
    __bf16* CAb  = (__bf16*)carve((size_t)BATCH * MM   * LPAD * 2); // 33.6 MB
    float*  OUT2 = (float*) carve((size_t)BATCH * MM   * DD2  * 4); // 41.9 MB

    // 1) bf16 query / key matrices (+ transposed padded keys), row norms
    k_build_q<<<(BATCH * MM * DD + 255) / 256, 256, 0, stream>>>(fg_in, Q);
    k_build_k<<<(BATCH * LPAD * DD2 + 255) / 256, 256, 0, stream>>>(bg_in, mask, UBr, UBt);
    k_qnorm  <<<(BATCH * MM   + 255) / 256, 256, 0, stream>>>(Q,   wwd);
    k_knorm  <<<(BATCH * LPAD + 255) / 256, 256, 0, stream>>>(UBr, k1d);

    // 2) GEMM1: DS1[m][n] = ||k||^2 + ||q||^2 - 2 q.k   (M=4096, N=1024, K=576)
    gemm_tn_wmma<<<dim3(MM / 128, LPAD / 128, BATCH), 256, 0, stream>>>(
        Q, UBr, DS1, k1d, wwd,
        DD, DD, DD, LPAD,
        (size_t)MM * DD, (size_t)LPAD * DD, (size_t)MM * LPAD, 1);

    // 3) fused row softmax (mean/std -> -tanh -> softmax), bf16 out, zero-padded
    k_softmax<<<BATCH * MM, 256, 0, stream>>>(DS1, CAb);

    // 4) GEMM2: OUT2[m][e] = sum_l CA[m][l] * key[l][e]  (M=4096, N=640, K=1024)
    gemm_tn_wmma<<<dim3(MM / 128, DD2 / 128, BATCH), 256, 0, stream>>>(
        CAb, UBt, OUT2, nullptr, nullptr,
        LPAD, LPAD, LPAD, DD2,
        (size_t)MM * LPAD, (size_t)DD2 * LPAD, (size_t)MM * DD2, 0);

    // 5) deconv-gather + mask blend + 1x1 fuse conv + ELU
    k_final<<<BATCH * HH, 256, 0, stream>>>(bg_in, mask, OUT2, fuse_w, fuse_b, out);
}
